// PSRNetwork_35115652612209
// MI455X (gfx1250) — matrix operand order repaired and checked by reading, not compiled, not run on hardware
//
#include <hip/hip_runtime.h>
#include <cstring>

// ---------------------------------------------------------------------------
// CDNA5 (gfx1250) GVP-GNN forward.  wave32, WMMA bf16 16x16x32, f32 accum.
// Weights pre-converted to bf16, transposed [N, Kpad] so B-fragments are one
// aligned 32B vector load; activations staged fp32 in LDS with zero-padded K
// so A-fragments are straight-line ds_load_b64 pairs. No per-element guards.
// ---------------------------------------------------------------------------

typedef __attribute__((ext_vector_type(16))) __bf16 v16bf;
typedef __attribute__((ext_vector_type(8)))  float  v8f;

#define NN   4096
#define NPGC 1024
#define KNBR 30
#define EE   (NN * KNBR)

// ---------------------------- WMMA helpers ---------------------------------

__device__ __forceinline__ v8f wmma_bf16(v16bf a, v16bf b, v8f c) {
  return __builtin_amdgcn_wmma_f32_16x16x32_bf16(false, a, false, b, (short)0, c,
                                                 false, false);
}

// A fragment (16x32 bf16) from fp32 row-major LDS tile; K range must be
// zero-padded by the caller.  Frag elem j: k = 16*half + j,
// i.e. contiguous pairs -> 8 x ds_load_b64.
__device__ __forceinline__ v16bf lda_lds(const float* base, int ld, int k0,
                                         int lane) {
  int m = lane & 15, half = lane >> 4;
  const float* row = base + m * ld + k0 + (half << 4);
  v16bf a;
#pragma unroll
  for (int jp = 0; jp < 8; ++jp) {
    int k = jp << 1;
    float2 x = *(const float2*)(row + k);
    a[2 * jp]     = (__bf16)x.x;
    a[2 * jp + 1] = (__bf16)x.y;
  }
  return a;
}

// B fragment (32x16 bf16) from pre-transposed bf16 weights [Nw][Kpad].
// Frag elem j: k = k0 + 16*half + j (contiguous) -> one aligned 32B load.
__device__ __forceinline__ v16bf ldb_bf16(const __bf16* wt, int Kpad, int k0,
                                          int n0, int lane) {
  int n = lane & 15, half = lane >> 4;
  return *(const v16bf*)(wt + (n0 + n) * Kpad + k0 + (half << 4));
}

__device__ __forceinline__ float wave_sum(float x) {
#pragma unroll
  for (int o = 16; o > 0; o >>= 1) x += __shfl_xor(x, o, 32);
  return x;
}

// ----------------------- weight convert (fp32->bf16^T) ----------------------

__global__ void wconv_kernel(const float* __restrict__ w, __bf16* __restrict__ wt,
                             int Kw, int Nw, int Kpad) {
  int t = blockIdx.x * blockDim.x + threadIdx.x;
  if (t >= Nw * Kpad) return;
  int n = t / Kpad, k = t - n * Kpad;
  float x = (k < Kw) ? w[(long long)k * Nw + n] : 0.0f;
  wt[t] = (__bf16)x;
}

// ------------------------------- kNN ---------------------------------------

__global__ void knn_kernel(const float* __restrict__ pos, int* __restrict__ idx) {
  int i = blockIdx.x * blockDim.x + threadIdx.x;
  if (i >= NN) return;
  int base = (i / NPGC) * NPGC;
  float px = pos[i * 3], py = pos[i * 3 + 1], pz = pos[i * 3 + 2];
  float prev_d = -1.0f; int prev_j = -1;
  for (int k = 0; k < KNBR; ++k) {
    float best_d = 3.4e38f; int best_j = 0x7fffffff;
    for (int j = 0; j < NPGC; ++j) {
      int jj = base + j;
      if (jj == i) continue;
      float dx = px - pos[jj * 3], dy = py - pos[jj * 3 + 1], dz = pz - pos[jj * 3 + 2];
      float d2 = dx * dx + dy * dy + dz * dz;
      bool gt_prev = (d2 > prev_d) || (d2 == prev_d && jj > prev_j);
      bool lt_best = (d2 < best_d) || (d2 == best_d && jj < best_j);
      if (gt_prev && lt_best) { best_d = d2; best_j = jj; }
    }
    idx[(long long)i * KNBR + k] = best_j;
    prev_d = best_d; prev_j = best_j;
  }
}

// --------------------------- edge features ---------------------------------

__global__ void edge_feat_kernel(const float* __restrict__ pos,
                                 const int* __restrict__ idx,
                                 float* __restrict__ edge_s,
                                 float* __restrict__ edge_v) {
  int e = blockIdx.x * blockDim.x + threadIdx.x;
  if (e >= EE) return;
  int i = e / KNBR;
  int j = idx[e];
  float d = (float)(i - j);
  float ex = pos[i * 3] - pos[j * 3];
  float ey = pos[i * 3 + 1] - pos[j * 3 + 1];
  float ez = pos[i * 3 + 2] - pos[j * 3 + 2];
  float D = sqrtf(ex * ex + ey * ey + ez * ez);
  for (int q = 0; q < 16; ++q) {
    float mu = 20.0f * (float)q / 15.0f;
    float z = (D - mu) * (16.0f / 20.0f);
    edge_s[(long long)e * 32 + q] = expf(-z * z);
  }
  const float kln = -0.57564627324851142f;  // -ln(10000)/16
  for (int t = 0; t < 8; ++t) {
    float ang = d * expf(kln * (float)(2 * t));
    edge_s[(long long)e * 32 + 16 + t] = cosf(ang);
    edge_s[(long long)e * 32 + 24 + t] = sinf(ang);
  }
  float inv = (D > 0.0f) ? (1.0f / D) : 0.0f;
  edge_v[e]                 = ex * inv;
  edge_v[(long long)EE + e] = ey * inv;
  edge_v[2LL * EE + e]      = ez * inv;
}

// ------------------------------ fused GVP -----------------------------------
// s_out = act( [s_parts..., ||v@wh||] @ ws + bs ), v_out = gate((v@wh) @ wv)
// One block = 16 rows (M always a multiple of 16); 4 waves do WMMA tiles.

struct GvpArgs {
  int M, K, vi, h, si, so, vo, act, n_sp, n_vp;
  const int* nbr;
  const float* sp[3]; int sw[3]; int sld[3]; int smode[3]; int sroff[3];
  const float* vp[3]; int vw[3]; int vmode[3]; int vroff[3];
  long long vrs[3], vcs[3], vxs[3];
  const __bf16 *wh, *ws, *wv;
  const float* bs;
  float* s_out;  float* v_out;  long long vo_ps;
};

__device__ __forceinline__ int gvp_maprow(int r, int mode, int roff,
                                          const int* nbr, int K) {
  int g = r + roff;
  if (mode == 1) return nbr[g];
  if (mode == 2) return g / K;
  return g;
}

__global__ __launch_bounds__(128) void gvp_kernel(GvpArgs a) {
  // 13824 floats = 54 KB static LDS.
  // region0 [0,9216): SV [3][16][96] (ph1-2)  /  SA [16][576] (ph3-4)
  //                   /  SVO [3][16][64] (ph5)   (disjoint lifetimes)
  // SVH [9216,13824): [3][16][96]
  __shared__ __align__(16) float smem[13824];
  float* SA  = smem;
  float* SV  = smem;
  float* SVO = smem;
  float* SVH = smem + 9216;
  const int tid = threadIdx.x, lane = tid & 31, wid = tid >> 5;
  const int r0 = blockIdx.x * 16;

  // ---- phase 1: gather v inputs into SV [c][m][96]; zero K pads of SV & SVH
  {
    int cb = 0;
    for (int p = 0; p < a.n_vp; ++p) {
      int w = a.vw[p], tot = 16 * 3 * w;
      for (int t = tid; t < tot; t += 128) {
        int m = t / (3 * w); int rem = t - m * 3 * w; int c = rem / w; int j = rem - c * w;
        int sr = gvp_maprow(r0 + m, a.vmode[p], a.vroff[p], a.nbr, a.K);
        SV[(c * 16 + m) * 96 + (cb + j)] =
            a.vp[p][(long long)sr * a.vrs[p] + (long long)j * a.vcs[p] +
                    (long long)c * a.vxs[p]];
      }
      cb += w;
    }
    int pv = 96 - a.vi;
    for (int t = tid; t < 48 * pv; t += 128) {
      int cm = t / pv, j = t - cm * pv;
      SV[cm * 96 + a.vi + j] = 0.0f;
    }
    int ph = 96 - a.h;
    for (int t = tid; t < 48 * ph; t += 128) {
      int cm = t / ph, j = t - cm * ph;
      SVH[cm * 96 + a.h + j] = 0.0f;
    }
  }
  __syncthreads();

  // ---- phase 2: v_h[c] = v[c] @ wh
  {
    int nht = a.h >> 4;
    int whKp = (a.vi + 31) & ~31;
    for (int t = wid; t < 3 * nht; t += 4) {
      int c = t / nht, n0 = (t - c * nht) << 4;
      v8f acc = {};
      for (int k0 = 0; k0 < a.vi; k0 += 32) {
        v16bf af = lda_lds(&SV[c * 16 * 96], 96, k0, lane);
        v16bf bf = ldb_bf16(a.wh, whKp, k0, n0, lane);
        acc = wmma_bf16(af, bf, acc);
      }
      int n = lane & 15, half = lane >> 4;
#pragma unroll
      for (int i = 0; i < 8; ++i)
        SVH[(c * 16 + (half * 8 + i)) * 96 + (n0 + n)] = acc[i];
    }
  }
  __syncthreads();  // SV dead; SA region usable

  // ---- phase 3: SA[:, si..si+h) = ||v_h|| ; SA[:, 0..si) = s parts ; pad
  {
    for (int t = tid; t < 16 * a.h; t += 128) {
      int m = t / a.h, j = t - m * a.h;
      float x0 = SVH[(0 * 16 + m) * 96 + j];
      float x1 = SVH[(1 * 16 + m) * 96 + j];
      float x2 = SVH[(2 * 16 + m) * 96 + j];
      SA[m * 576 + a.si + j] = sqrtf(fmaxf(x0 * x0 + x1 * x1 + x2 * x2, 1e-8f));
    }
    int cb = 0;
    for (int p = 0; p < a.n_sp; ++p) {
      int w = a.sw[p];
      for (int t = tid; t < 16 * w; t += 128) {
        int m = t / w, j = t - m * w;
        int sr = gvp_maprow(r0 + m, a.smode[p], a.sroff[p], a.nbr, a.K);
        SA[m * 576 + cb + j] = a.sp[p][(long long)sr * a.sld[p] + j];
      }
      cb += w;
    }
    int Ks = a.si + a.h;
    int pad = ((Ks + 31) & ~31) - Ks;
    for (int t = tid; t < 16 * pad; t += 128) {
      int m = t / pad, j = t - m * pad;
      SA[m * 576 + Ks + j] = 0.0f;
    }
  }
  __syncthreads();

  // ---- phase 4: s_out = SA @ ws + bs  (+ReLU)
  {
    int Ks = a.si + a.h;
    int wsKp = (Ks + 31) & ~31;
    int nso = a.so >> 4;
    for (int t = wid; t < nso; t += 4) {
      int n0 = t << 4;
      v8f acc = {};
      for (int k0 = 0; k0 < Ks; k0 += 32) {
        v16bf af = lda_lds(SA, 576, k0, lane);
        v16bf bf = ldb_bf16(a.ws, wsKp, k0, n0, lane);
        acc = wmma_bf16(af, bf, acc);
      }
      int n = lane & 15, half = lane >> 4;
      float bias = a.bs[n0 + n];
#pragma unroll
      for (int i = 0; i < 8; ++i) {
        int gr = r0 + half * 8 + i;
        float x = acc[i] + bias;
        if (a.act) x = fmaxf(x, 0.0f);
        a.s_out[(long long)gr * a.so + (n0 + n)] = x;
      }
    }
  }

  // ---- phase 5: v_out = v_h @ wv, gated by sigmoid(||v_out||)
  if (a.vo > 0) {
    __syncthreads();  // SA dead; SVO region usable
    int nvt = a.vo >> 4;
    int wvKp = (a.h + 31) & ~31;
    for (int t = wid; t < 3 * nvt; t += 4) {
      int c = t / nvt, n0 = (t - c * nvt) << 4;
      v8f acc = {};
      for (int k0 = 0; k0 < a.h; k0 += 32) {
        v16bf af = lda_lds(&SVH[c * 16 * 96], 96, k0, lane);
        v16bf bf = ldb_bf16(a.wv, wvKp, k0, n0, lane);
        acc = wmma_bf16(af, bf, acc);
      }
      int n = lane & 15, half = lane >> 4;
#pragma unroll
      for (int i = 0; i < 8; ++i)
        SVO[(c * 16 + (half * 8 + i)) * 64 + (n0 + n)] = acc[i];
    }
    __syncthreads();
    for (int t = tid; t < 16 * a.vo; t += 128) {
      int m = t / a.vo, j = t - m * a.vo;
      int gr = r0 + m;
      float x0 = SVO[(0 * 16 + m) * 64 + j];
      float x1 = SVO[(1 * 16 + m) * 64 + j];
      float x2 = SVO[(2 * 16 + m) * 64 + j];
      float sc = 1.0f;
      if (a.act) {
        float nrm = sqrtf(fmaxf(x0 * x0 + x1 * x1 + x2 * x2, 1e-8f));
        sc = 1.0f / (1.0f + expf(-nrm));
      }
      long long off = (long long)gr * a.vo + j;
      a.v_out[off]               = x0 * sc;
      a.v_out[a.vo_ps + off]     = x1 * sc;
      a.v_out[2 * a.vo_ps + off] = x2 * sc;
    }
  }
}

// ----------------------------- LayerNorm ------------------------------------

__global__ __launch_bounds__(128) void ln_kernel(
    float* s, const float* res_s, const float* __restrict__ g,
    const float* __restrict__ b, float* v, const float* res_v,
    int M, int F, int W, long long vps, long long rvps) {
  int row = blockIdx.x * 4 + (threadIdx.x >> 5);
  int lane = threadIdx.x & 31;
  if (row >= M) return;
  float sum = 0.0f, sq = 0.0f;
  for (int f = lane; f < F; f += 32) {
    long long o = (long long)row * F + f;
    float x = s[o];
    if (res_s) x += res_s[o];
    s[o] = x;
    sum += x; sq += x * x;
  }
  sum = wave_sum(sum); sq = wave_sum(sq);
  float mu = sum / (float)F;
  float var = fmaxf(sq / (float)F - mu * mu, 0.0f);
  float inv = rsqrtf(var + 1e-5f);
  for (int f = lane; f < F; f += 32) {
    long long o = (long long)row * F + f;
    s[o] = (s[o] - mu) * inv * g[f] + b[f];
  }
  if (W > 0) {
    float acc = 0.0f;
    for (int t = lane; t < 3 * W; t += 32) {
      int c = t / W, j = t - c * W;
      long long o = (long long)c * vps + (long long)row * W + j;
      float x = v[o];
      if (res_v) x += res_v[(long long)c * rvps + (long long)row * W + j];
      v[o] = x;
      acc += x * x;
    }
    acc = wave_sum(acc);
    float vinv = rsqrtf(fmaxf(acc / (float)W, 1e-8f));
    for (int t = lane; t < 3 * W; t += 32) {
      int c = t / W, j = t - c * W;
      v[(long long)c * vps + (long long)row * W + j] *= vinv;
    }
  }
}

// --------------------------- mean aggregation --------------------------------

__global__ void agg_kernel(const float* __restrict__ ms, const float* __restrict__ mv,
                           float* __restrict__ ds, float* __restrict__ dv,
                           int n0, int nn, long long mvps) {
  int t = blockIdx.x * blockDim.x + threadIdx.x;
  int tot = nn * (128 + 96);
  if (t >= tot) return;
  if (t < nn * 128) {
    int l = t >> 7, f = t & 127;
    float s = 0.0f;
    const float* p = ms + (long long)l * KNBR * 128 + f;
    for (int k = 0; k < KNBR; ++k) s += p[k * 128];
    ds[(long long)(n0 + l) * 128 + f] = s * (1.0f / (float)KNBR);
  } else {
    int t2 = t - nn * 128;
    int l = t2 / 96; int rem = t2 - l * 96; int c = rem >> 5, j = rem & 31;
    float s = 0.0f;
    const float* p = mv + (long long)c * mvps + (long long)l * KNBR * 32 + j;
    for (int k = 0; k < KNBR; ++k) s += p[k * 32];
    dv[(long long)c * NN * 32 + (long long)(n0 + l) * 32 + j] = s * (1.0f / (float)KNBR);
  }
}

// --------------------------- pooling + dense ---------------------------------

__global__ __launch_bounds__(128) void pool_dense_kernel(
    const float* __restrict__ outs, const float* __restrict__ w1,
    const float* __restrict__ b1, const float* __restrict__ w2,
    const float* __restrict__ b2, float* __restrict__ out) {
  __shared__ float pool[128];
  __shared__ float hb[128];
  int b = blockIdx.x, f = threadIdx.x;
  float s = 0.0f;
  const float* p = outs + (long long)b * NPGC * 128 + f;
  for (int n = 0; n < NPGC; ++n) s += p[(long long)n * 128];
  pool[f] = s / (float)NPGC;
  __syncthreads();
  float acc = b1[f];
  for (int i = 0; i < 128; ++i) acc += pool[i] * w1[i * 128 + f];
  hb[f] = fmaxf(acc, 0.0f) * w2[f];
  __syncthreads();
  if (f == 0) {
    float t = 0.0f;
    for (int i = 0; i < 128; ++i) t += hb[i];
    out[b] = t + b2[0] + 0.5f;
  }
}

// ------------------------------ host side ------------------------------------

static GvpArgs gvp_make(int M) {
  GvpArgs a; memset(&a, 0, sizeof(a)); a.M = M; a.K = KNBR; return a;
}
static void gvp_add_s(GvpArgs& a, const float* p, int w, int ld, int mode, int roff) {
  int i = a.n_sp++; a.sp[i] = p; a.sw[i] = w; a.sld[i] = ld;
  a.smode[i] = mode; a.sroff[i] = roff; a.si += w;
}
static void gvp_add_v(GvpArgs& a, const float* p, int w, long long rs, long long cs,
                      long long xs, int mode, int roff) {
  int i = a.n_vp++; a.vp[i] = p; a.vw[i] = w; a.vrs[i] = rs; a.vcs[i] = cs;
  a.vxs[i] = xs; a.vmode[i] = mode; a.vroff[i] = roff; a.vi += w;
}
static void gvp_run(hipStream_t st, GvpArgs& a, int h, int so, int vo, int act,
                    const __bf16* wh, const __bf16* ws, const float* bs,
                    const __bf16* wv, float* sout, float* vout, long long vops,
                    const int* nbr) {
  a.h = h; a.so = so; a.vo = vo; a.act = act;
  a.wh = wh; a.ws = ws; a.bs = bs; a.wv = wv;
  a.s_out = sout; a.v_out = vout; a.vo_ps = vops; a.nbr = nbr;
  int tiles = (a.M + 15) / 16;
  gvp_kernel<<<dim3(tiles), dim3(128), 0, st>>>(a);
}

extern "C" void kernel_launch(void* const* d_in, const int* in_sizes, int n_in,
                              void* d_out, int out_size, void* d_ws, size_t ws_size,
                              hipStream_t stream) {
  (void)in_sizes; (void)n_in; (void)out_size; (void)ws_size;
  const int EC = KNBR * 1024, ECn = 1024, NCH = 4;   // edge chunks (node-aligned)
  auto F = [&](int i) { return (const float*)d_in[i]; };

  const float* node_s = F(0);
  const float* node_v = F(1);     // [N,3,3] row/ch/coord
  const float* pos_CA = F(2);
  // d_in[3]=pos_C, d_in[4]=pos_N, d_in[5]=batch_ids: unused (rotation-conjugation
  // is an exact identity for channel-mixing + norm-gated nets; graphs contiguous).

  // ---- workspace (bump allocator) ----
  char* wp = (char*)d_ws;
  auto alloc = [&](size_t bytes) -> char* {
    char* r = wp; wp += (bytes + 255) & ~(size_t)255; return r;
  };
  int*   idx    = (int*)  alloc((size_t)NN * KNBR * 4);
  float* edge_s = (float*)alloc((size_t)EE * 32 * 4);
  float* edge_v = (float*)alloc((size_t)3 * EE * 4);
  float* hs     = (float*)alloc((size_t)NN * 128 * 4);
  float* hv     = (float*)alloc((size_t)3 * NN * 32 * 4);
  float* es     = (float*)alloc((size_t)EE * 64 * 4);
  float* ev     = (float*)alloc((size_t)3 * EE * 16 * 4);
  float* dsb    = (float*)alloc((size_t)NN * 128 * 4);
  float* dvb    = (float*)alloc((size_t)3 * NN * 32 * 4);
  float* ts1    = (float*)alloc((size_t)NN * 512 * 4);
  float* tv1    = (float*)alloc((size_t)3 * NN * 64 * 4);
  float* ts2    = (float*)alloc((size_t)NN * 512 * 4);
  float* tv2    = (float*)alloc((size_t)3 * NN * 64 * 4);
  float* csa    = (float*)alloc((size_t)EC * 128 * 4);
  float* cva    = (float*)alloc((size_t)3 * EC * 32 * 4);
  float* csb    = (float*)alloc((size_t)EC * 128 * 4);
  float* cvb    = (float*)alloc((size_t)3 * EC * 32 * 4);

  // ---- pre-convert all GVP weights: fp32 [K,N] -> bf16 [N][Kpad] ----
  __bf16* wt[103] = {};
  auto conv = [&](int i, int Kw, int Nw) {
    int Kpad = (Kw + 31) & ~31;
    __bf16* dst = (__bf16*)alloc((size_t)Nw * Kpad * 2);
    int tot = Nw * Kpad;
    wconv_kernel<<<dim3((tot + 255) / 256), dim3(256), 0, stream>>>(F(i), dst,
                                                                    Kw, Nw, Kpad);
    wt[i] = dst;
  };
  conv(6, 3, 32);    conv(7, 58, 128);   conv(9, 32, 32);     // W_node.g1
  conv(12, 32, 32);  conv(13, 160, 128); conv(15, 32, 32);    // W_node.g2
  conv(18, 1, 16);   conv(19, 48, 64);   conv(21, 16, 16);    // W_edge.g
  for (int L = 0; L < 3; ++L) {
    int wb = 22 + L * 24;
    conv(wb + 0, 80, 80);  conv(wb + 1, 400, 128); conv(wb + 3, 80, 32);   // msg0
    conv(wb + 4, 32, 32);  conv(wb + 5, 160, 128); conv(wb + 7, 32, 32);   // msg1
    conv(wb + 8, 32, 32);  conv(wb + 9, 160, 128); conv(wb + 11, 32, 32);  // msg2
    conv(wb + 14, 32, 64); conv(wb + 15, 192, 512); conv(wb + 17, 64, 64); // ff0
    conv(wb + 18, 64, 64); conv(wb + 19, 576, 128); conv(wb + 21, 64, 32); // ff1
  }
  conv(96, 32, 32);  conv(97, 160, 128);                      // W_out.g

  const long long HVP = (long long)NN * 32;   // hv plane stride
  const long long EVP = (long long)EE * 16;   // ev plane stride
  const long long CVP = (long long)EC * 32;   // chunk v plane stride

  // ---- graph + edge features ----
  knn_kernel<<<dim3((NN + 127) / 128), dim3(128), 0, stream>>>(pos_CA, idx);
  edge_feat_kernel<<<dim3((EE + 127) / 128), dim3(128), 0, stream>>>(pos_CA, idx,
                                                                     edge_s, edge_v);

  // ---- W_node: g1 -> ln -> g2 ----
  {
    GvpArgs a = gvp_make(NN);
    gvp_add_s(a, node_s, 26, 26, 0, 0);
    gvp_add_v(a, node_v, 3, 9, 3, 1, 0, 0);
    gvp_run(stream, a, 32, 128, 32, 0, wt[6], wt[7], F(8), wt[9],
            ts1, tv1, (long long)NN * 32, idx);
  }
  ln_kernel<<<dim3((NN + 3) / 4), dim3(128), 0, stream>>>(
      ts1, nullptr, F(10), F(11), tv1, nullptr, NN, 128, 32, (long long)NN * 32, 0);
  {
    GvpArgs a = gvp_make(NN);
    gvp_add_s(a, ts1, 128, 128, 0, 0);
    gvp_add_v(a, tv1, 32, 32, 1, (long long)NN * 32, 0, 0);
    gvp_run(stream, a, 32, 128, 32, 0, wt[12], wt[13], F(14), wt[15],
            hs, hv, HVP, idx);
  }

  // ---- W_edge: ln -> g ----
  ln_kernel<<<dim3((EE + 3) / 4), dim3(128), 0, stream>>>(
      edge_s, nullptr, F(16), F(17), edge_v, nullptr, EE, 32, 1, (long long)EE, 0);
  {
    GvpArgs a = gvp_make(EE);
    gvp_add_s(a, edge_s, 32, 32, 0, 0);
    gvp_add_v(a, edge_v, 1, 1, 1, (long long)EE, 0, 0);
    gvp_run(stream, a, 16, 64, 16, 0, wt[18], wt[19], F(20), wt[21],
            es, ev, EVP, idx);
  }

  // ---- message-passing layers ----
  for (int L = 0; L < 3; ++L) {
    int wb = 22 + L * 24;
    for (int c = 0; c < NCH; ++c) {
      int e0 = c * EC;
      {  // msg GVP0: s=[hs[nbr], es, hs[center]] (320), v=[hv[nbr], ev, hv[ctr]] (80)
        GvpArgs a = gvp_make(EC);
        gvp_add_s(a, hs, 128, 128, 1, e0);
        gvp_add_s(a, es, 64, 64, 0, e0);
        gvp_add_s(a, hs, 128, 128, 2, e0);
        gvp_add_v(a, hv, 32, 32, 1, HVP, 1, e0);
        gvp_add_v(a, ev, 16, 16, 1, EVP, 0, e0);
        gvp_add_v(a, hv, 32, 32, 1, HVP, 2, e0);
        gvp_run(stream, a, 80, 128, 32, 1, wt[wb + 0], wt[wb + 1], F(wb + 2),
                wt[wb + 3], csa, cva, CVP, idx);
      }
      {  // msg GVP1
        GvpArgs a = gvp_make(EC);
        gvp_add_s(a, csa, 128, 128, 0, 0);
        gvp_add_v(a, cva, 32, 32, 1, CVP, 0, 0);
        gvp_run(stream, a, 32, 128, 32, 1, wt[wb + 4], wt[wb + 5], F(wb + 6),
                wt[wb + 7], csb, cvb, CVP, idx);
      }
      {  // msg GVP2 (no act)
        GvpArgs a = gvp_make(EC);
        gvp_add_s(a, csb, 128, 128, 0, 0);
        gvp_add_v(a, cvb, 32, 32, 1, CVP, 0, 0);
        gvp_run(stream, a, 32, 128, 32, 0, wt[wb + 8], wt[wb + 9], F(wb + 10),
                wt[wb + 11], csa, cva, CVP, idx);
      }
      int tot = ECn * (128 + 96);
      agg_kernel<<<dim3((tot + 255) / 256), dim3(256), 0, stream>>>(
          csa, cva, dsb, dvb, c * ECn, ECn, CVP);
    }
    // residual + ln1
    ln_kernel<<<dim3((NN + 3) / 4), dim3(128), 0, stream>>>(
        hs, dsb, F(wb + 12), F(wb + 13), hv, dvb, NN, 128, 32, HVP, HVP);
    // feed-forward: ff0 (act) -> ff1
    {
      GvpArgs a = gvp_make(NN);
      gvp_add_s(a, hs, 128, 128, 0, 0);
      gvp_add_v(a, hv, 32, 32, 1, HVP, 0, 0);
      gvp_run(stream, a, 64, 512, 64, 1, wt[wb + 14], wt[wb + 15], F(wb + 16),
              wt[wb + 17], ts1, tv1, (long long)NN * 64, idx);
    }
    {
      GvpArgs a = gvp_make(NN);
      gvp_add_s(a, ts1, 512, 512, 0, 0);
      gvp_add_v(a, tv1, 64, 64, 1, (long long)NN * 64, 0, 0);
      gvp_run(stream, a, 64, 128, 32, 0, wt[wb + 18], wt[wb + 19], F(wb + 20),
              wt[wb + 21], ts2, tv2, HVP, idx);
    }
    // residual + ln2
    ln_kernel<<<dim3((NN + 3) / 4), dim3(128), 0, stream>>>(
        hs, ts2, F(wb + 22), F(wb + 23), hv, tv2, NN, 128, 32, HVP, HVP);
  }

  // ---- W_out: ln -> g (scalar only) ----
  ln_kernel<<<dim3((NN + 3) / 4), dim3(128), 0, stream>>>(
      hs, nullptr, F(94), F(95), hv, nullptr, NN, 128, 32, HVP, 0);
  {
    GvpArgs a = gvp_make(NN);
    gvp_add_s(a, hs, 128, 128, 0, 0);
    gvp_add_v(a, hv, 32, 32, 1, HVP, 0, 0);
    gvp_run(stream, a, 32, 128, 0, 0, wt[96], wt[97], F(98), nullptr,
            ts1, nullptr, 0, idx);
  }

  // ---- pool + dense head ----
  pool_dense_kernel<<<dim3(4), dim3(128), 0, stream>>>(
      ts1, F(99), F(100), F(101), F(102), (float*)d_out);
}
